// VNSelfAttention_49855980372168
// MI455X (gfx1250) — matrix operand order repaired and checked by reading, not compiled, not run on hardware
//
#include <hip/hip_runtime.h>
#include <stdint.h>

#define BDIM 8
#define CDIM 256
#define NDIM 1024
#define HDIM 8
#define DD   96            // flattened head dim (32 * 3)
#define TN   (3 * NDIM)    // 3072 columns in the channel GEMMs

typedef __attribute__((ext_vector_type(16))) _Float16 v16h;
typedef __attribute__((ext_vector_type(8)))  _Float16 v8h;
typedef __attribute__((ext_vector_type(8)))  float    v8f;

// --- WMMA fragment helpers (wave32, 16x16x32 f16) -------------------------
// A (16x32): row = lane&15; lanes 0-15 hold K {0-7,16-23}, lanes 16-31 {8-15,24-31}
// B (32x16): col = lane&15; same K order as A
__device__ __forceinline__ v8f wmma_f16(v16h a, v16h b, v8f c) {
  return __builtin_amdgcn_wmma_f32_16x16x32_f16(false, a, false, b,
                                                (short)0, c, false, false);
}
__device__ __forceinline__ v16h load_a_frag(const _Float16* rowp, int lane) {
  const int hi8 = (lane >> 4) << 3;
  v8h lo = *(const v8h*)(rowp + hi8);
  v8h hi = *(const v8h*)(rowp + 16 + hi8);
  return __builtin_shufflevector(lo, hi, 0, 1, 2, 3, 4, 5, 6, 7,
                                 8, 9, 10, 11, 12, 13, 14, 15);
}
__device__ __forceinline__ v16h load_b_frag(const _Float16* kp, int lane) {
  const int hi8 = (lane >> 4) << 3;
  v8h lo = *(const v8h*)(kp + hi8);
  v8h hi = *(const v8h*)(kp + 16 + hi8);
  return __builtin_shufflevector(lo, hi, 0, 1, 2, 3, 4, 5, 6, 7,
                                 8, 9, 10, 11, 12, 13, 14, 15);
}

// ---------------------------------------------------------------------------
// Kernel 0a: x[b,c,(dvec,n)] f32 -> xT[b, j=dvec*N+n, c] f16, tiled transpose.
// Global->LDS staging uses the CDNA5 async-to-LDS engine (ASYNCcnt).
// ---------------------------------------------------------------------------
__global__ __launch_bounds__(256)
void vn_x_transpose(const float* __restrict__ x, _Float16* __restrict__ xT)
{
  __shared__ float tile[64 * 65];          // +1 pad: conflict-free transpose
  const int tid = threadIdx.x;
  const int j0 = blockIdx.x * 64;          // TN/64 = 48
  const int c0 = blockIdx.y * 64;          // C/64  = 4
  const int b  = blockIdx.z;
  const float* xb = x + ((size_t)b * CDIM + c0) * TN + j0;

  // ---- load 64x64 f32 tile, coalesced along j, async into LDS ----
  const int col  = tid & 63;
  const int row0 = tid >> 6;               // 0..3
#pragma unroll
  for (int i = 0; i < 16; ++i) {
    const int row = row0 + i * 4;
    const uint32_t ldsoff = (uint32_t)(uintptr_t)&tile[row * 65 + col];
    const float* gp = xb + (size_t)row * TN + col;
    asm volatile("global_load_async_to_lds_b32 %0, %1, off"
                 :: "v"(ldsoff), "v"(gp) : "memory");
  }
  asm volatile("s_wait_asynccnt 0x0" ::: "memory");
  __syncthreads();

  // ---- write transposed, f16, coalesced along c ----
  const int j  = tid >> 2;                 // 0..63
  const int cq = (tid & 3) << 4;           // 0,16,32,48
  v16h out;
#pragma unroll
  for (int e = 0; e < 16; ++e)
    out[e] = (_Float16)tile[(cq + e) * 65 + j];
  *(v16h*)(xT + ((size_t)b * TN + j0 + j) * CDIM + c0 + cq) = out;
}

// ---------------------------------------------------------------------------
// Kernel 0b: convert the four [C,C] f32 weights to f16 (re-read ~100x later).
// ---------------------------------------------------------------------------
__global__ __launch_bounds__(256)
void vn_w_convert(const float* __restrict__ wq, const float* __restrict__ wk,
                  const float* __restrict__ wv, const float* __restrict__ wp,
                  _Float16* __restrict__ hq, _Float16* __restrict__ hk,
                  _Float16* __restrict__ hv, _Float16* __restrict__ hp)
{
  const int i = blockIdx.x * 256 + threadIdx.x;
  hq[i] = (_Float16)wq[i];
  hk[i] = (_Float16)wk[i];
  hv[i] = (_Float16)wv[i];
  hp[i] = (_Float16)wp[i];
}

// ---------------------------------------------------------------------------
// Kernel 1: fused Q/K/V projection, 16(o) x 32(j) tile per wave:
// per K-chunk 5 fragment loads -> 6 WMMAs (B shared by 3 weights, A by 2 cols).
// q,k -> [B,H,N,96]; v -> transposed [B,H,96,N].
// ---------------------------------------------------------------------------
__global__ __launch_bounds__(32)
void vn_qkv_proj(const _Float16* __restrict__ xT,
                 const _Float16* __restrict__ Whq,
                 const _Float16* __restrict__ Whk,
                 const _Float16* __restrict__ Whv,
                 _Float16* __restrict__ qh,
                 _Float16* __restrict__ kh,
                 _Float16* __restrict__ vT)
{
  const int lane = threadIdx.x & 31;
  const int j0 = blockIdx.x * 32;
  const int o0 = blockIdx.y * 16;
  const int b  = blockIdx.z;

  const int arow = o0 + (lane & 15);
  const _Float16* xrow0 = xT + ((size_t)b * TN + j0 + (lane & 15)) * CDIM;
  const _Float16* xrow1 = xrow0 + (size_t)16 * CDIM;

  v8f aq[2] = {}, ak[2] = {}, av[2] = {};
  for (int k0 = 0; k0 < CDIM; k0 += 32) {
    const v16h fb0 = load_b_frag(xrow0 + k0, lane);
    const v16h fb1 = load_b_frag(xrow1 + k0, lane);
    const v16h fq = load_a_frag(Whq + arow * CDIM + k0, lane);
    const v16h fk = load_a_frag(Whk + arow * CDIM + k0, lane);
    const v16h fv = load_a_frag(Whv + arow * CDIM + k0, lane);
    aq[0] = wmma_f16(fq, fb0, aq[0]);
    aq[1] = wmma_f16(fq, fb1, aq[1]);
    ak[0] = wmma_f16(fk, fb0, ak[0]);
    ak[1] = wmma_f16(fk, fb1, ak[1]);
    av[0] = wmma_f16(fv, fb0, av[0]);
    av[1] = wmma_f16(fv, fb1, av[1]);
  }

#pragma unroll
  for (int p = 0; p < 2; ++p) {
    const int jj   = j0 + p * 16 + (lane & 15);
    const int dvec = jj >> 10;             // / NDIM
    const int n    = jj & (NDIM - 1);
#pragma unroll
    for (int r = 0; r < 8; ++r) {
      const int o   = o0 + r + ((lane >> 4) << 3);
      const int h   = o >> 5;
      const int hdi = o & 31;
      const size_t bh = (size_t)b * HDIM + h;
      const int d = hdi * 3 + dvec;
      qh[(bh * NDIM + n) * DD + d] = (_Float16)aq[p][r];
      kh[(bh * NDIM + n) * DD + d] = (_Float16)ak[p][r];
      vT[(bh * DD + d) * NDIM + n] = (_Float16)av[p][r];
    }
  }
}

// ---------------------------------------------------------------------------
// Kernel 2: masked softmax attention for one (b, h, 16-row q tile).
// Q fragments hoisted; scores in 32KB LDS; P@V reads transposed V so all
// B-fragments are two 16B row loads. Output -> yt[b, dvec, n, c] f16.
// ---------------------------------------------------------------------------
__global__ __launch_bounds__(32)
void vn_attention(const _Float16* __restrict__ qh,
                  const _Float16* __restrict__ kh,
                  const _Float16* __restrict__ vT,
                  const uint8_t* __restrict__ mask,
                  _Float16* __restrict__ yt)
{
  __shared__ _Float16 Sh[16 * NDIM];       // 32 KB scores
  __shared__ float rowmax[16];
  __shared__ float rowsum[16];

  const int lane = threadIdx.x & 31;
  const int nt = blockIdx.x;
  const int h  = blockIdx.y;
  const int b  = blockIdx.z;
  const size_t bh = (size_t)b * HDIM + h;
  const _Float16* Q = qh + bh * NDIM * DD;
  const _Float16* K = kh + bh * NDIM * DD;
  const _Float16* V = vT + bh * DD * NDIM; // [96, 1024]
  const uint8_t* mk = mask + (size_t)b * NDIM;
  const float scale = 0.10206207262f;      // 1/sqrt(96)

  // ---- hoisted Q fragments (invariant over key tiles) ----
  const int arow = nt * 16 + (lane & 15);
  v16h fq[3];
#pragma unroll
  for (int dci = 0; dci < 3; ++dci)
    fq[dci] = load_a_frag(Q + (size_t)arow * DD + dci * 32, lane);

  // ---- phase 1: scores into LDS ----
  for (int mt = 0; mt < NDIM / 16; ++mt) {
    const int mcol = mt * 16 + (lane & 15);
    const uint8_t masked = mk[mcol];
    v8f acc = {};
#pragma unroll
    for (int dci = 0; dci < 3; ++dci) {
      const v16h fb = load_b_frag(K + (size_t)mcol * DD + dci * 32, lane);
      acc = wmma_f16(fq[dci], fb, acc);
    }
#pragma unroll
    for (int r = 0; r < 8; ++r) {
      const int row = r + ((lane >> 4) << 3);
      float s = acc[r] * scale;
      if (masked) s = -30000.0f;           // padded key (f16-safe "-inf")
      Sh[row * NDIM + mcol] = (_Float16)s;
    }
  }
  __syncthreads();

  // ---- phase 2: row max & sum (vectorized LDS reads) ----
  {
    const int row  = lane & 15;
    const int half = lane >> 4;
    const _Float16* Srow = Sh + row * NDIM + half * (NDIM / 2);
    float mx = -1e30f;
    for (int m = 0; m < NDIM / 2; m += 8) {
      const v8h v = *(const v8h*)(Srow + m);
#pragma unroll
      for (int e = 0; e < 8; ++e) mx = fmaxf(mx, (float)v[e]);
    }
    mx = fmaxf(mx, __shfl_xor(mx, 16, 32));
    float sm = 0.0f;
    for (int m = 0; m < NDIM / 2; m += 8) {
      const v8h v = *(const v8h*)(Srow + m);
#pragma unroll
      for (int e = 0; e < 8; ++e) sm += __expf((float)v[e] - mx);
    }
    sm += __shfl_xor(sm, 16, 32);
    if (lane < 16) { rowmax[row] = mx; rowsum[row] = sm; }
  }
  __syncthreads();

  // ---- phase 3: O = exp(S - max) @ V ----
  const float rm = rowmax[lane & 15];
  const int hi8 = (lane >> 4) << 3;
  v8f O[6] = {};
  for (int m0 = 0; m0 < NDIM; m0 += 32) {
    const _Float16* Srow = Sh + (lane & 15) * NDIM + m0;
    const v8h slo = *(const v8h*)(Srow + hi8);
    const v8h shi = *(const v8h*)(Srow + 16 + hi8);
    v16h fp;
#pragma unroll
    for (int e = 0; e < 8; ++e) {
      fp[e]     = (_Float16)__expf((float)slo[e] - rm);
      fp[8 + e] = (_Float16)__expf((float)shi[e] - rm);
    }
#pragma unroll
    for (int t = 0; t < 6; ++t) {
      const v16h fb = load_b_frag(V + (size_t)(t * 16 + (lane & 15)) * NDIM + m0, lane);
      O[t] = wmma_f16(fp, fb, O[t]);       // A fragment reused 6x
    }
  }

  // ---- epilogue: divide by row sum, scatter into yt[b, dvec, n, c] ----
#pragma unroll
  for (int r = 0; r < 8; ++r) {
    const int row = r + ((lane >> 4) << 3);
    const int n = nt * 16 + row;
    const float inv = 1.0f / rowsum[row];
#pragma unroll
    for (int t = 0; t < 6; ++t) {
      const int d    = t * 16 + (lane & 15);
      const int dvec = d % 3;
      const int hdi  = d / 3;
      const int c    = h * 32 + hdi;
      yt[(((size_t)b * 3 + dvec) * NDIM + n) * CDIM + c] = (_Float16)(O[t][r] * inv);
    }
  }
}

// ---------------------------------------------------------------------------
// Kernel 3: output projection, 32(o) x 32(j) tile per wave:
// per K-chunk 4 fragment loads -> 4 WMMAs (each fragment reused twice).
// ---------------------------------------------------------------------------
__global__ __launch_bounds__(32)
void vn_out_proj(const _Float16* __restrict__ yt,
                 const _Float16* __restrict__ Whp,
                 float* __restrict__ out)
{
  const int lane = threadIdx.x & 31;
  const int j0 = blockIdx.x * 32;
  const int o0 = blockIdx.y * 32;
  const int b  = blockIdx.z;

  const int arow0 = o0 + (lane & 15);
  const _Float16* yrow0 = yt + ((size_t)b * TN + j0 + (lane & 15)) * CDIM;
  const _Float16* yrow1 = yrow0 + (size_t)16 * CDIM;

  v8f acc[2][2] = {};
  for (int k0 = 0; k0 < CDIM; k0 += 32) {
    const v16h fb0 = load_b_frag(yrow0 + k0, lane);
    const v16h fb1 = load_b_frag(yrow1 + k0, lane);
    const v16h fa0 = load_a_frag(Whp + arow0 * CDIM + k0, lane);
    const v16h fa1 = load_a_frag(Whp + (arow0 + 16) * CDIM + k0, lane);
    acc[0][0] = wmma_f16(fa0, fb0, acc[0][0]);
    acc[0][1] = wmma_f16(fa0, fb1, acc[0][1]);
    acc[1][0] = wmma_f16(fa1, fb0, acc[1][0]);
    acc[1][1] = wmma_f16(fa1, fb1, acc[1][1]);
  }
#pragma unroll
  for (int i = 0; i < 2; ++i) {
#pragma unroll
    for (int p = 0; p < 2; ++p) {
      const int jj = j0 + p * 16 + (lane & 15);
#pragma unroll
      for (int r = 0; r < 8; ++r) {
        const int o = o0 + i * 16 + r + ((lane >> 4) << 3);
        out[((size_t)b * CDIM + o) * TN + jj] = acc[i][p][r];
      }
    }
  }
}

// ---------------------------------------------------------------------------
extern "C" void kernel_launch(void* const* d_in, const int* in_sizes, int n_in,
                              void* d_out, int out_size, void* d_ws, size_t ws_size,
                              hipStream_t stream)
{
  (void)in_sizes; (void)n_in; (void)out_size; (void)ws_size;
  const float*   x    = (const float*)d_in[0];
  const float*   Wq   = (const float*)d_in[1];
  const float*   Wk   = (const float*)d_in[2];
  const float*   Wv   = (const float*)d_in[3];
  const float*   Wp   = (const float*)d_in[4];
  const uint8_t* mask = (const uint8_t*)d_in[5];   // jax bool -> 1 byte
  float* out = (float*)d_out;

  const size_t HEADS = (size_t)BDIM * HDIM * NDIM * DD;  // 6.29M halfs
  const size_t WSZ   = (size_t)CDIM * CDIM;              // 65536 halfs
  _Float16* xT  = (_Float16*)d_ws;      // [B, TN, C]
  _Float16* qh  = xT + (size_t)BDIM * TN * CDIM;
  _Float16* kh  = qh + HEADS;           // [B,H,N,96]
  _Float16* vT  = kh + HEADS;           // [B,H,96,N]
  _Float16* yt  = vT + HEADS;           // [B,3,N,C]
  _Float16* whq = yt + (size_t)BDIM * 3 * NDIM * CDIM;
  _Float16* whk = whq + WSZ;
  _Float16* whv = whk + WSZ;
  _Float16* whp = whv + WSZ;            // total ws: ~63.5 MB

  dim3 gT(TN / 64, CDIM / 64, BDIM);         // 48 x 4 x 8
  vn_x_transpose<<<gT, 256, 0, stream>>>(x, xT);

  vn_w_convert<<<dim3(CDIM * CDIM / 256), 256, 0, stream>>>(
      Wq, Wk, Wv, Wp, whq, whk, whv, whp);

  dim3 gQKV(TN / 32, CDIM / 16, BDIM);       // 96 x 16 x 8
  vn_qkv_proj<<<gQKV, 32, 0, stream>>>(xT, whq, whk, whv, qh, kh, vT);

  dim3 gAtt(NDIM / 16, HDIM, BDIM);          // 64 x 8 x 8
  vn_attention<<<gAtt, 32, 0, stream>>>(qh, kh, vT, mask, yt);

  dim3 gOut(TN / 32, CDIM / 32, BDIM);       // 96 x 8 x 8
  vn_out_proj<<<gOut, 32, 0, stream>>>(yt, whp, out);
}
